// LLAMA_87213605912788
// MI455X (gfx1250) — matrix operand order repaired and checked by reading, not compiled, not run on hardware
//
#include <hip/hip_runtime.h>
#include <hip/hip_bf16.h>

// ---------------------------------------------------------------------------
// LLaMA forward (4 layers, DIM=2048, SEQ=2048, 16 heads x 128, HIDDEN=5632,
// VOCAB=32000) for MI455X (gfx1250). All matmuls in bf16 WMMA w/ f32 accum.
// GEMM: double-buffered LDS, async global->LDS (CDNA5 ASYNCcnt path) for A,
// register-prefetched f32->bf16 transposed B tiles, optional in-block M loop
// for L2-friendly reuse of very large weight matrices (lm-head).
// ---------------------------------------------------------------------------

#define DIM     2048
#define SEQ     2048
#define NH      16
#define HD      128
#define HIDDEN  5632
#define VOCAB   32000
#define NLAYERS 4

#define USE_ASYNC_LDS 1

typedef __bf16 bf16;
typedef __attribute__((ext_vector_type(16))) __bf16 v16bf;
typedef __attribute__((ext_vector_type(8)))  __bf16 v8bf;
typedef __attribute__((ext_vector_type(8)))  float  v8f;

__device__ __forceinline__ v16bf concat8(v8bf lo, v8bf hi) {
    v16bf r;
#pragma unroll
    for (int i = 0; i < 8; ++i) { r[i] = lo[i]; r[i + 8] = hi[i]; }
    return r;
}

__device__ __forceinline__ v8f zero8() {
    v8f r;
#pragma unroll
    for (int i = 0; i < 8; ++i) r[i] = 0.f;
    return r;
}

// Generic shared pointers carry the LDS byte offset in their low 32 bits
__device__ __forceinline__ void async_copy_b128(void* lds_dst, const void* gsrc) {
#if USE_ASYNC_LDS
    uint32_t loff = (uint32_t)(uintptr_t)lds_dst;
    asm volatile("global_load_async_to_lds_b128 %0, %1, off"
                 :: "v"(loff), "v"(gsrc)
                 : "memory");
#else
    *(v8bf*)lds_dst = *(const v8bf*)gsrc;
#endif
}

__device__ __forceinline__ void wait_async() {
#if USE_ASYNC_LDS
    asm volatile("s_wait_asynccnt 0x0" ::: "memory");
#endif
}

// ---------------------------------------------------------------------------
// Generic GEMM: C(MxN,f32) = A(MxK,bf16) @ W(KxN,f32->bf16) [+ R(MxN,f32)]
// Block: 256 threads = 8 waves; tile 128(M) x 128(N); K step 32.
// Wave grid 4(M) x 2(N): each wave computes 32x64 = 2x4 WMMA fragments.
// MLOOP m-tiles are processed per block against the same weight column-block
// so big weight panels stay L2-resident across the M sweep.
// ---------------------------------------------------------------------------
template <int ADD_RES, int MLOOP>
__global__ __launch_bounds__(256) void gemm_kernel(
    const bf16* __restrict__ A, const float* __restrict__ W,
    float* __restrict__ C, const float* __restrict__ R,
    int M, int N, int K) {
    __shared__ bf16 lA[2][128 * 32];   // [row][k], row-major
    __shared__ bf16 lB[2][128 * 32];   // [n][k], transposed

    const int t       = threadIdx.x;
    const int wv      = t >> 5;
    const int lane    = t & 31;
    const int lid     = lane & 15;
    const int halfsel = lane >> 4;
    const int wm      = wv & 3;     // M offset wm*32
    const int wn      = wv >> 2;    // N offset wn*64
    const int n0      = blockIdx.x * 128;
    const int NK      = K >> 5;

    // thread's slice of the cooperative loads
    const int arow0 = (t + 0)   >> 2, ac0 = (t + 0)   & 3;
    const int arow1 = (t + 256) >> 2, ac1 = (t + 256) & 3;

    for (int mt = 0; mt < MLOOP; ++mt) {
        const int m0 = (blockIdx.y * MLOOP + mt) * 128;

        float4 breg[4];
        auto issue_a = [&](int buf, int kk) {
            async_copy_b128(&lA[buf][arow0 * 32 + ac0 * 8],
                            &A[(size_t)(m0 + arow0) * K + kk + ac0 * 8]);
            async_copy_b128(&lA[buf][arow1 * 32 + ac1 * 8],
                            &A[(size_t)(m0 + arow1) * K + kk + ac1 * 8]);
        };
        auto issue_b = [&](int kk) {
#pragma unroll
            for (int j = 0; j < 4; ++j) {
                int idx = t + 256 * j;
                int kq  = idx >> 5;
                int n4  = idx & 31;
                breg[j] = *(const float4*)&W[(size_t)(kk + kq) * N + n0 + n4 * 4];
            }
        };
        auto store_b = [&](int buf) {
#pragma unroll
            for (int j = 0; j < 4; ++j) {
                int idx = t + 256 * j;
                int kq  = idx >> 5;
                int n4  = idx & 31;
                lB[buf][(n4 * 4 + 0) * 32 + kq] = (bf16)breg[j].x;
                lB[buf][(n4 * 4 + 1) * 32 + kq] = (bf16)breg[j].y;
                lB[buf][(n4 * 4 + 2) * 32 + kq] = (bf16)breg[j].z;
                lB[buf][(n4 * 4 + 3) * 32 + kq] = (bf16)breg[j].w;
            }
        };

        v8f acc[2][4];
#pragma unroll
        for (int mi = 0; mi < 2; ++mi)
#pragma unroll
            for (int ni = 0; ni < 4; ++ni) acc[mi][ni] = zero8();

        // prologue: tile 0
        issue_a(0, 0);
        issue_b(0);
        store_b(0);
        wait_async();
        __syncthreads();

        for (int kt = 0; kt < NK; ++kt) {
            const int cur  = kt & 1;
            const int nxt  = cur ^ 1;
            const bool more = (kt + 1 < NK);
            if (more) {           // overlap next-tile traffic with WMMAs
                issue_a(nxt, (kt + 1) * 32);
                issue_b((kt + 1) * 32);
            }

            v16bf a[2], b[4];
#pragma unroll
            for (int mi = 0; mi < 2; ++mi) {
                int r  = wm * 32 + mi * 16 + lid;
                int k0 = halfsel ? 8 : 0;  // A layout: K {0-7,16-23}/{8-15,24-31}
                a[mi] = concat8(*(const v8bf*)&lA[cur][r * 32 + k0],
                                *(const v8bf*)&lA[cur][r * 32 + k0 + 16]);
            }
#pragma unroll
            for (int ni = 0; ni < 4; ++ni) {
                int n  = wn * 64 + ni * 16 + lid;
                int k0 = halfsel ? 8 : 0;  // B layout: K {0-7,16-23}/{8-15,24-31}
                b[ni] = concat8(*(const v8bf*)&lB[cur][n * 32 + k0],
                                *(const v8bf*)&lB[cur][n * 32 + k0 + 16]);
            }
#pragma unroll
            for (int mi = 0; mi < 2; ++mi)
#pragma unroll
                for (int ni = 0; ni < 4; ++ni)
                    acc[mi][ni] = __builtin_amdgcn_wmma_f32_16x16x32_bf16(
                        false, a[mi], false, b[ni], (short)0, acc[mi][ni],
                        false, false);

            if (more) store_b(nxt);
            wait_async();         // next A tile resident in LDS
            __syncthreads();
        }

        // epilogue: C layout: VGPR g, lane L -> row g + 8*(L>=16), col L%16
#pragma unroll
        for (int mi = 0; mi < 2; ++mi)
#pragma unroll
            for (int ni = 0; ni < 4; ++ni)
#pragma unroll
                for (int g = 0; g < 8; ++g) {
                    int row = m0 + wm * 32 + mi * 16 + g + 8 * halfsel;
                    int col = n0 + wn * 64 + ni * 16 + lid;
                    float v = acc[mi][ni][g];
                    if (ADD_RES) v += R[(size_t)row * N + col];
                    C[(size_t)row * N + col] = v;
                }
        if (MLOOP > 1) __syncthreads();   // LDS reuse safety between m-tiles
    }
}

// ---------------------------------------------------------------------------
// Flash attention: one wave per (16-row Q block, head). Causal.
// ---------------------------------------------------------------------------
__global__ __launch_bounds__(32) void attn_kernel(
    const bf16* __restrict__ qb, const bf16* __restrict__ kb,
    const bf16* __restrict__ vt, bf16* __restrict__ ao) {
    __shared__ bf16 psh[16 * 32];
    const int lane    = threadIdx.x;
    const int lid     = lane & 15;
    const int halfsel = lane >> 4;
    const int r0      = blockIdx.x * 16;
    const int hh      = blockIdx.y;

    // Q A-fragments: rows r0..r0+15, head-dim 128 = 4 frags of K=32
    v16bf aq[4];
    {
        const bf16* qrow = qb + (size_t)(r0 + lid) * DIM + hh * HD;
#pragma unroll
        for (int f = 0; f < 4; ++f) {
            int k0 = f * 32 + (halfsel ? 8 : 0);
            aq[f] = concat8(*(const v8bf*)&qrow[k0],
                            *(const v8bf*)&qrow[k0 + 16]);
        }
    }

    v8f o[8];
#pragma unroll
    for (int d = 0; d < 8; ++d) o[d] = zero8();
    float mrow[8], lrow[8];
#pragma unroll
    for (int g = 0; g < 8; ++g) { mrow[g] = -1e30f; lrow[g] = 0.f; }

    const float sc = 0.0883883476483184f;   // 1/sqrt(128)

    for (int j0 = 0; j0 <= r0 + 15; j0 += 32) {
        // S = Q @ K^T : two 16x16 subtiles over this 32-col block
        v8f s01[2];
        s01[0] = zero8(); s01[1] = zero8();
#pragma unroll
        for (int sub = 0; sub < 2; ++sub) {
            int col = j0 + sub * 16 + lid;
            if (col > SEQ - 1) col = SEQ - 1;          // clamp (masked later)
            const bf16* krow = kb + (size_t)col * DIM + hh * HD;
#pragma unroll
            for (int f = 0; f < 4; ++f) {
                int k0 = f * 32 + (halfsel ? 8 : 0);
                v16bf bk = concat8(*(const v8bf*)&krow[k0],
                                   *(const v8bf*)&krow[k0 + 16]);
                s01[sub] = __builtin_amdgcn_wmma_f32_16x16x32_bf16(
                    false, aq[f], false, bk, (short)0, s01[sub], false, false);
            }
        }
        // online softmax (rows live as C-layout element g, half halfsel)
#pragma unroll
        for (int g = 0; g < 8; ++g) {
            int rrow = r0 + g + 8 * halfsel;
            float mx = mrow[g];
#pragma unroll
            for (int sub = 0; sub < 2; ++sub) {
                int c   = j0 + sub * 16 + lid;
                float v = s01[sub][g] * sc;
                if (c > rrow || c >= SEQ) v = -1e30f;  // causal mask
                s01[sub][g] = v;
                mx = fmaxf(mx, v);
            }
#pragma unroll
            for (int off = 1; off < 16; off <<= 1)
                mx = fmaxf(mx, __shfl_xor(mx, off, 16));
            float al = __expf(mrow[g] - mx);
            mrow[g] = mx;
            float ps = 0.f;
#pragma unroll
            for (int sub = 0; sub < 2; ++sub) {
                float p = __expf(s01[sub][g] - mx);
                ps += p;
                psh[(g + 8 * halfsel) * 32 + sub * 16 + lid] = (bf16)p;
            }
#pragma unroll
            for (int off = 1; off < 16; off <<= 1)
                ps += __shfl_xor(ps, off, 16);
            lrow[g] = lrow[g] * al + ps;
#pragma unroll
            for (int d = 0; d < 8; ++d) o[d][g] *= al;
        }
        __syncthreads();
        // reshape P (C-layout) -> A-fragment via LDS
        v16bf pa;
        {
            int k0 = halfsel ? 8 : 0;
            pa = concat8(*(const v8bf*)&psh[lid * 32 + k0],
                         *(const v8bf*)&psh[lid * 32 + k0 + 16]);
        }
        // O += P @ V  (V pre-transposed: vt[h*HD+d][s], contiguous in s)
#pragma unroll
        for (int d = 0; d < 8; ++d) {
            const bf16* vrow =
                vt + (size_t)(hh * HD + d * 16 + lid) * SEQ + j0 + (halfsel ? 8 : 0);
            v16bf bv = concat8(*(const v8bf*)&vrow[0], *(const v8bf*)&vrow[16]);
            o[d] = __builtin_amdgcn_wmma_f32_16x16x32_bf16(
                false, pa, false, bv, (short)0, o[d], false, false);
        }
        __syncthreads();
    }

    // normalize + store (bf16, feeds the wo GEMM)
#pragma unroll
    for (int g = 0; g < 8; ++g) {
        float inv = 1.f / lrow[g];
        int row = r0 + g + 8 * halfsel;
#pragma unroll
        for (int d = 0; d < 8; ++d)
            ao[(size_t)row * DIM + hh * HD + d * 16 + lid] = (bf16)(o[d][g] * inv);
    }
}

// ---------------------------------------------------------------------------
// Elementwise kernels
// ---------------------------------------------------------------------------
__global__ __launch_bounds__(256) void embed_kernel(
    const int* __restrict__ tokens, const float* __restrict__ emb,
    float* __restrict__ h) {
    int s = blockIdx.x;
    const float* e = emb + (size_t)tokens[s] * DIM;
    float* hr = h + (size_t)s * DIM;
    for (int i = threadIdx.x; i < DIM; i += 256) hr[i] = e[i];
}

__global__ __launch_bounds__(256) void rmsnorm_kernel(
    const float* __restrict__ x, const float* __restrict__ w,
    bf16* __restrict__ out) {
    __shared__ float red[256];
    int row = blockIdx.x, t = threadIdx.x;
    const float* xr = x + (size_t)row * DIM;
    float s = 0.f;
    for (int i = t; i < DIM; i += 256) { float v = xr[i]; s += v * v; }
    red[t] = s;
    __syncthreads();
    for (int st = 128; st > 0; st >>= 1) {
        if (t < st) red[t] += red[t + st];
        __syncthreads();
    }
    float inv = rsqrtf(red[0] / (float)DIM + 1e-6f);
    for (int i = t; i < DIM; i += 256)
        out[(size_t)row * DIM + i] = (bf16)(xr[i] * inv * w[i]);
}

__global__ __launch_bounds__(64) void rope_kernel(
    const float* __restrict__ q, const float* __restrict__ k,
    bf16* __restrict__ qb, bf16* __restrict__ kb) {
    int s = blockIdx.x, hh = blockIdx.y, d = threadIdx.x;   // d in 0..63
    float invf = __expf(-logf(10000.f) * (float)(2 * d) / (float)HD);
    float ang = (float)s * invf;
    float sn, cs;
    __sincosf(ang, &sn, &cs);
    size_t base = (size_t)s * DIM + hh * HD + d;
    float q0 = q[base], q1 = q[base + 64];
    qb[base]      = (bf16)(q0 * cs - q1 * sn);
    qb[base + 64] = (bf16)(q1 * cs + q0 * sn);
    float k0 = k[base], k1 = k[base + 64];
    kb[base]      = (bf16)(k0 * cs - k1 * sn);
    kb[base + 64] = (bf16)(k1 * cs + k0 * sn);
}

__global__ __launch_bounds__(256) void transpose_v_kernel(
    const float* __restrict__ v, bf16* __restrict__ vt) {
    __shared__ float tile[32][33];
    int s0 = blockIdx.x * 32, c0 = blockIdx.y * 32;
    int x = threadIdx.x, y = threadIdx.y;   // block (32,8)
#pragma unroll
    for (int k = 0; k < 4; ++k)
        tile[y + 8 * k][x] = v[(size_t)(s0 + y + 8 * k) * DIM + c0 + x];
    __syncthreads();
#pragma unroll
    for (int k = 0; k < 4; ++k)
        vt[(size_t)(c0 + y + 8 * k) * SEQ + s0 + x] = (bf16)tile[x][y + 8 * k];
}

__global__ __launch_bounds__(256) void silumul_kernel(
    const float* __restrict__ a, const float* __restrict__ b,
    bf16* __restrict__ g, int n) {
    int i = blockIdx.x * 256 + threadIdx.x;
    if (i < n) {
        float x = a[i];
        float s = x / (1.f + __expf(-x));
        g[i] = (bf16)(s * b[i]);
    }
}

// ---------------------------------------------------------------------------
// Host launch
// ---------------------------------------------------------------------------
static inline size_t alignup(size_t x) { return (x + 255) & ~(size_t)255; }

extern "C" void kernel_launch(void* const* d_in, const int* in_sizes, int n_in,
                              void* d_out, int out_size, void* d_ws, size_t ws_size,
                              hipStream_t stream) {
    const int*   tokens      = (const int*)d_in[0];
    const float* emb         = (const float*)d_in[1];
    const float* wq          = (const float*)d_in[2];
    const float* wk          = (const float*)d_in[3];
    const float* wv          = (const float*)d_in[4];
    const float* wo          = (const float*)d_in[5];
    const float* w1          = (const float*)d_in[6];
    const float* w2          = (const float*)d_in[7];
    const float* w3          = (const float*)d_in[8];
    const float* attn_norm_w = (const float*)d_in[9];
    const float* ffn_norm_w  = (const float*)d_in[10];
    const float* final_norm  = (const float*)d_in[11];
    const float* out_w       = (const float*)d_in[12];

    // workspace carve
    const size_t SD  = (size_t)SEQ * DIM;
    const size_t SH  = (size_t)SEQ * HIDDEN;
    char* base = (char*)d_ws;
    size_t off = 0;
    auto alloc = [&](size_t bytes) -> void* {
        void* p = base + off;
        off = alignup(off + bytes);
        return p;
    };
    float* h    = (float*)alloc(SD * 4);                       // residual stream
    bf16*  xb   = (bf16*) alloc(SD * 2);                       // bf16 activations
    size_t bigf_bytes = (3 * SD * 4 > 2 * SH * 4) ? 3 * SD * 4 : 2 * SH * 4;
    float* bigf = (float*)alloc(bigf_bytes);                   // q,k,v | h1,h3
    size_t bigb_bytes = (3 * SD * 2 > SH * 2) ? 3 * SD * 2 : SH * 2;
    bf16*  bigb = (bf16*) alloc(bigb_bytes + 256);             // qb,kb,vt | gb

    float* qf = bigf;
    float* kf = bigf + SD;
    float* vf = bigf + 2 * SD;
    float* h1 = bigf;
    float* h3 = bigf + SH;
    bf16*  qb = bigb;
    bf16*  kb = bigb + SD;
    bf16*  vt = bigb + 2 * SD;
    bf16*  gb = bigb;

    const dim3 gD(DIM / 128, SEQ / 128);        // 2048-wide GEMMs
    const dim3 gH(HIDDEN / 128, SEQ / 128);     // FFN up GEMMs
    const dim3 gV(VOCAB / 128, SEQ / 128 / 8);  // lm-head, 8 m-tiles per block

    embed_kernel<<<SEQ, 256, 0, stream>>>(tokens, emb, h);

    for (int l = 0; l < NLAYERS; ++l) {
        const float* wq_l = wq + (size_t)l * DIM * DIM;
        const float* wk_l = wk + (size_t)l * DIM * DIM;
        const float* wv_l = wv + (size_t)l * DIM * DIM;
        const float* wo_l = wo + (size_t)l * DIM * DIM;
        const float* w1_l = w1 + (size_t)l * DIM * HIDDEN;
        const float* w2_l = w2 + (size_t)l * HIDDEN * DIM;
        const float* w3_l = w3 + (size_t)l * DIM * HIDDEN;

        rmsnorm_kernel<<<SEQ, 256, 0, stream>>>(h, attn_norm_w + (size_t)l * DIM, xb);
        gemm_kernel<0, 1><<<gD, 256, 0, stream>>>(xb, wq_l, qf, nullptr, SEQ, DIM, DIM);
        gemm_kernel<0, 1><<<gD, 256, 0, stream>>>(xb, wk_l, kf, nullptr, SEQ, DIM, DIM);
        gemm_kernel<0, 1><<<gD, 256, 0, stream>>>(xb, wv_l, vf, nullptr, SEQ, DIM, DIM);
        rope_kernel<<<dim3(SEQ, NH), 64, 0, stream>>>(qf, kf, qb, kb);
        transpose_v_kernel<<<dim3(SEQ / 32, DIM / 32), dim3(32, 8), 0, stream>>>(vf, vt);
        attn_kernel<<<dim3(SEQ / 16, NH), 32, 0, stream>>>(qb, kb, vt, xb);
        gemm_kernel<1, 1><<<gD, 256, 0, stream>>>(xb, wo_l, h, h, SEQ, DIM, DIM);

        rmsnorm_kernel<<<SEQ, 256, 0, stream>>>(h, ffn_norm_w + (size_t)l * DIM, xb);
        gemm_kernel<0, 1><<<gH, 256, 0, stream>>>(xb, w1_l, h1, nullptr, SEQ, HIDDEN, DIM);
        gemm_kernel<0, 1><<<gH, 256, 0, stream>>>(xb, w3_l, h3, nullptr, SEQ, HIDDEN, DIM);
        int n = (int)SH;
        silumul_kernel<<<(n + 255) / 256, 256, 0, stream>>>(h1, h3, gb, n);
        gemm_kernel<1, 1><<<gD, 256, 0, stream>>>(gb, w2_l, h, h, SEQ, DIM, HIDDEN);
    }

    rmsnorm_kernel<<<SEQ, 256, 0, stream>>>(h, final_norm, xb);
    gemm_kernel<0, 8><<<gV, 256, 0, stream>>>(xb, out_w, (float*)d_out, nullptr,
                                              SEQ, VOCAB, DIM);
}
